// GNN_61684320305464
// MI455X (gfx1250) — compile-verified, matrix operand order checked by reading, not run on hardware
//
#include <hip/hip_runtime.h>

// ---------------------------------------------------------------------------
// CDNA5 (gfx1250) GNN forward.  All large GEMMs run on v_wmma_f32_16x16x32_bf16
// (wave32, 16x16x32 bf16 -> f32).  Compute: ~0.43 TFLOP of GEMM; memory:
// ~0.7 GB workspace traffic -> matrix-core bound, bf16 inputs / f32 accum.
//
// d_in flattening assumption (setup_inputs dict insertion order, params
// expanded in written order):
//  0 x_s_raw  1 x_t_raw  2 edge_attr  3 x_u
//  4 enc_s_W 5 enc_s_b 6 enc_t_W 7 enc_t_b
//  8 eW1 9 eb1 10 eW2 11 eb2 12 e_g 13 e_b
// 14 mW1 15 mb1 16 mW2 17 mb2
// 18 uW1 19 ub1 20 uW2 21 ub2 22 u_g 23 u_b
// 24 gW1 25 gb1 26 gW2 27 gb2 28 g_g 29 g_b
// 30 src 31 tgt
// d_out = [x_s (NS*128) | x_t (NT*128) | u (128)] fp32.
// Workspace requirement ~705 MB.
// ---------------------------------------------------------------------------

typedef __bf16 bf16_t;
typedef __attribute__((ext_vector_type(16))) __bf16 v16bf;
typedef __attribute__((ext_vector_type(8)))  __bf16 v8bf;
typedef __attribute__((ext_vector_type(8)))  float  v8f;

#define F      128
#define NS_N   50000
#define NT_N   256
#define E_N    200000
#define NLAYER 2
#define EPSV   1e-5f

#define CEILDIV(a,b) (((a)+(b)-1)/(b))

// ---------------------------------------------------------------------------
// WMMA fragment load: row-major [rows, ld] bf16.  For lane L (wave32):
//   m/n = L&15, group g = L>>4.  Elements 0..7 -> k = kk + g*8 + i,
//   elements 8..15 -> k = kk + 16 + g*8 + (i-8).  Two 16B contiguous loads.
// Same mapping used for A (rows = M) and B (rows = N, weights pre-transposed
// to [N,K] so B columns are contiguous in K).
// ---------------------------------------------------------------------------
__device__ __forceinline__ v16bf load_frag(const bf16_t* __restrict__ base,
                                           int ld, int row, int kk, int lane) {
  const int r = row + (lane & 15);
  const int g = lane >> 4;
  const bf16_t* p = base + (size_t)r * (size_t)ld + kk + g * 8;
  v8bf lo = *(const v8bf*)(p);
  v8bf hi = *(const v8bf*)(p + 16);
  return __builtin_shufflevector(lo, hi, 0,1,2,3,4,5,6,7,8,9,10,11,12,13,14,15);
}

// Generic bf16 GEMM: out[M,N] = act(A[M,K] @ W[K,N] + bias)
// A row-major bf16, Wt = W transposed, [N,K] row-major bf16.
// Block: 256 threads = 8 waves, block tile 64(M) x 128(N), wave tile 32x32.
// Requires N % 128 == 0 (gridDim.y = N/128), M padded to 64 in the A buffer.
template<bool RELU, int OUTBF>
__global__ __launch_bounds__(256)
void wmma_gemm_bf16(const bf16_t* __restrict__ A, const bf16_t* __restrict__ Wt,
                    const float* __restrict__ bias, void* __restrict__ out,
                    int M, int N, int K) {
  const int lane  = threadIdx.x & 31;
  const int wave  = threadIdx.x >> 5;          // 0..7
  const int wn    = wave & 3;                  // 4 waves across N
  const int wm    = wave >> 2;                 // 2 waves down M
  const int rBase = blockIdx.x * 64 + wm * 32;
  const int cBase = blockIdx.y * 128 + wn * 32;

  v8f acc00 = {}; v8f acc01 = {}; v8f acc10 = {}; v8f acc11 = {};

  for (int kk = 0; kk < K; kk += 32) {
    if (kk + 32 < K) {
      __builtin_prefetch((const void*)(A + (size_t)(rBase + (lane & 15)) * K + kk + 32), 0, 0);
    }
    v16bf a0 = load_frag(A,  K, rBase,      kk, lane);
    v16bf a1 = load_frag(A,  K, rBase + 16, kk, lane);
    v16bf b0 = load_frag(Wt, K, cBase,      kk, lane);
    v16bf b1 = load_frag(Wt, K, cBase + 16, kk, lane);
    acc00 = __builtin_amdgcn_wmma_f32_16x16x32_bf16(false, a0, false, b0, (short)0, acc00, false, false);
    acc01 = __builtin_amdgcn_wmma_f32_16x16x32_bf16(false, a0, false, b1, (short)0, acc01, false, false);
    acc10 = __builtin_amdgcn_wmma_f32_16x16x32_bf16(false, a1, false, b0, (short)0, acc10, false, false);
    acc11 = __builtin_amdgcn_wmma_f32_16x16x32_bf16(false, a1, false, b1, (short)0, acc11, false, false);
  }

  // C/D layout: element r of lane L -> m = r + 8*(L>>4), n = L&15 (16x16 tile)
  const int mo = (lane >> 4) * 8;
  const int no = lane & 15;
  float*  outf = (float*)out;
  bf16_t* outb = (bf16_t*)out;

  auto store_tile = [&](const v8f& acc, int tm, int tn) {
#pragma unroll
    for (int r = 0; r < 8; ++r) {
      int m = rBase + tm + mo + r;
      int n = cBase + tn + no;
      if (m < M) {
        float v = acc[r] + bias[n];
        if (RELU) v = fmaxf(v, 0.0f);
        if (OUTBF) outb[(size_t)m * N + n] = (bf16_t)v;
        else       outf[(size_t)m * N + n] = v;
      }
    }
  };
  store_tile(acc00,  0,  0); store_tile(acc01,  0, 16);
  store_tile(acc10, 16,  0); store_tile(acc11, 16, 16);
}

// ---------------------------------------------------------------------------
// Support kernels
// ---------------------------------------------------------------------------

// fp32 W[K,N] -> bf16 Wt[N,K]
__global__ void k_transpose_bf16(const float* __restrict__ W, bf16_t* __restrict__ Wt,
                                 int K, int N) {
  int idx = blockIdx.x * blockDim.x + threadIdx.x;
  if (idx >= K * N) return;
  int k = idx / N, n = idx - k * N;
  Wt[(size_t)n * K + k] = (bf16_t)W[idx];
}

__global__ void k_enc_s(const float* __restrict__ raw, const float* __restrict__ W,
                        const float* __restrict__ b, float* __restrict__ xs) {
  int idx = blockIdx.x * blockDim.x + threadIdx.x;
  if (idx >= NS_N * F) return;
  int i = idx >> 7, j = idx & 127;
  xs[idx] = fmaxf(raw[i] * W[j] + b[j], 0.0f);
}

__global__ void k_enc_t(const float* __restrict__ raw, const float* __restrict__ W,
                        const float* __restrict__ b, float* __restrict__ xt) {
  int idx = blockIdx.x * blockDim.x + threadIdx.x;
  if (idx >= NT_N * F) return;
  int i = idx >> 7, j = idx & 127;
  xt[idx] = fmaxf(raw[2 * i] * W[j] + raw[2 * i + 1] * W[F + j] + b[j], 0.0f);
}

__global__ void k_copy_u(const float* __restrict__ xu, float* __restrict__ u) {
  int t = threadIdx.x;
  if (t < F) u[t] = xu[t];
}

__global__ void k_count(const int* __restrict__ src, float* __restrict__ cnt) {
  int e = blockIdx.x * blockDim.x + threadIdx.x;
  if (e < E_N) atomicAdd(&cnt[src[e]], 1.0f);
}

// H[e, 0:512] = [x_s[src[e]], x_t[tgt[e]], edge_attr[e], u]  (bf16)
__global__ void k_edge_concat(const float* __restrict__ xs, const float* __restrict__ xt,
                              const float* __restrict__ ea, const float* __restrict__ u,
                              const int* __restrict__ src, const int* __restrict__ tgt,
                              bf16_t* __restrict__ H) {
  int idx = blockIdx.x * blockDim.x + threadIdx.x;
  if (idx >= E_N * 512) return;
  int e = idx >> 9, j = idx & 511;
  float v;
  if (j < 128)      v = xs[(size_t)src[e] * F + j];
  else if (j < 256) v = xt[(size_t)tgt[e] * F + (j - 128)];
  else if (j < 384) v = ea[(size_t)e * F + (j - 256)];
  else              v = u[j - 384];
  H[(size_t)e * 512 + j] = (bf16_t)v;
}

// Mc[e, 0:256] = [xnode[idx[e]], e_bn[e]]  (bf16)
__global__ void k_msg_concat(const float* __restrict__ xnode, const int* __restrict__ idx,
                             const float* __restrict__ e_bn, bf16_t* __restrict__ Mc) {
  int t = blockIdx.x * blockDim.x + threadIdx.x;
  if (t >= E_N * 256) return;
  int e = t >> 8, j = t & 255;
  float v = (j < 128) ? xnode[(size_t)idx[e] * F + j]
                      : e_bn[(size_t)e * F + (j - 128)];
  Mc[(size_t)e * 256 + j] = (bf16_t)v;
}

__global__ void k_scatter_add(const float* __restrict__ msg, const int* __restrict__ idx,
                              float* __restrict__ agg) {
  int t = blockIdx.x * blockDim.x + threadIdx.x;
  if (t >= E_N * F) return;
  int e = t >> 7, j = t & 127;
  atomicAdd(&agg[(size_t)idx[e] * F + j], msg[t]);
}

// U[i, 0:256] = [x[i], agg[i] (optionally / max(cnt,1))]  (bf16)
__global__ void k_upd_concat(const float* __restrict__ x, const float* __restrict__ agg,
                             const float* __restrict__ cnt, int M, bf16_t* __restrict__ U) {
  int t = blockIdx.x * blockDim.x + threadIdx.x;
  if (t >= M * 256) return;
  int i = t >> 8, j = t & 255;
  float v;
  if (j < 128) v = x[(size_t)i * F + j];
  else {
    v = agg[(size_t)i * F + (j - 128)];
    if (cnt) v /= fmaxf(cnt[i], 1.0f);
  }
  U[(size_t)i * 256 + j] = (bf16_t)v;
}

// Column sum / sum-of-squares over M rows of [M,128] fp32; blockDim.x == 128.
__global__ void k_colstats(const float* __restrict__ X, int M,
                           float* __restrict__ sum, float* __restrict__ sumsq) {
  int col = threadIdx.x;
  float s = 0.f, s2 = 0.f;
  for (int r = blockIdx.x; r < M; r += gridDim.x) {
    float v = X[(size_t)r * F + col];
    s += v; s2 += v * v;
  }
  atomicAdd(&sum[col], s);
  if (sumsq) atomicAdd(&sumsq[col], s2);
}

__global__ void k_bn_final(const float* __restrict__ sum, const float* __restrict__ sumsq,
                           const float* __restrict__ g, const float* __restrict__ b,
                           int M, float* __restrict__ scale, float* __restrict__ shift) {
  int c = threadIdx.x;
  if (c >= F) return;
  float mu  = sum[c] / (float)M;
  float var = sumsq[c] / (float)M - mu * mu;
  float sc  = g[c] * rsqrtf(var + EPSV);
  scale[c] = sc;
  shift[c] = b[c] - mu * sc;
}

__global__ void k_bn_apply(const float* __restrict__ X, const float* __restrict__ scale,
                           const float* __restrict__ shift, int M, float* __restrict__ Y) {
  int t = blockIdx.x * blockDim.x + threadIdx.x;
  if (t >= M * F) return;
  int j = t & 127;
  Y[t] = X[t] * scale[j] + shift[j];
}

// GlobalModel: hg = [u, mean(x_s), mean(x_t)] (3f=384); Linear->ReLU->Linear->LN.
__global__ __launch_bounds__(384)
void k_global(const float* __restrict__ u, const float* __restrict__ sum_s,
              const float* __restrict__ sum_t,
              const float* __restrict__ W1, const float* __restrict__ b1,
              const float* __restrict__ W2, const float* __restrict__ b2,
              const float* __restrict__ lg, const float* __restrict__ lb,
              float* __restrict__ u_out) {
  __shared__ float hg[384];
  __shared__ float hid[384];
  __shared__ float outv[128];
  __shared__ float red[2];
  int t = threadIdx.x;
  if (t < 128)      hg[t] = u[t];
  else if (t < 256) hg[t] = sum_s[t - 128] * (1.0f / (float)NS_N);
  else              hg[t] = sum_t[t - 256] * (1.0f / (float)NT_N);
  __syncthreads();
  float a1 = b1[t];
  for (int k = 0; k < 384; ++k) a1 += hg[k] * W1[k * 384 + t];
  hid[t] = fmaxf(a1, 0.0f);
  __syncthreads();
  if (t < 128) {
    float a2 = b2[t];
    for (int k = 0; k < 384; ++k) a2 += hid[k] * W2[k * 128 + t];
    outv[t] = a2;
  }
  __syncthreads();
  if (t == 0) {
    float s = 0.f, s2 = 0.f;
    for (int c = 0; c < 128; ++c) { s += outv[c]; s2 += outv[c] * outv[c]; }
    float mu = s / 128.0f;
    red[0] = mu;
    red[1] = rsqrtf(s2 / 128.0f - mu * mu + EPSV);
  }
  __syncthreads();
  if (t < 128) u_out[t] = (outv[t] - red[0]) * red[1] * lg[t] + lb[t];
}

// ---------------------------------------------------------------------------
// Host orchestration
// ---------------------------------------------------------------------------
extern "C" void kernel_launch(void* const* d_in, const int* in_sizes, int n_in,
                              void* d_out, int out_size, void* d_ws, size_t ws_size,
                              hipStream_t stream) {
  (void)in_sizes; (void)n_in; (void)out_size; (void)ws_size;

  const float* x_s_raw = (const float*)d_in[0];
  const float* x_t_raw = (const float*)d_in[1];
  const float* edge_attr = (const float*)d_in[2];
  const float* x_u   = (const float*)d_in[3];
  const float* enc_s_W = (const float*)d_in[4];
  const float* enc_s_b = (const float*)d_in[5];
  const float* enc_t_W = (const float*)d_in[6];
  const float* enc_t_b = (const float*)d_in[7];
  const float* eW1 = (const float*)d_in[8];
  const float* eb1 = (const float*)d_in[9];
  const float* eW2 = (const float*)d_in[10];
  const float* eb2 = (const float*)d_in[11];
  const float* e_g = (const float*)d_in[12];
  const float* e_b = (const float*)d_in[13];
  const float* mW1 = (const float*)d_in[14];
  const float* mb1 = (const float*)d_in[15];
  const float* mW2 = (const float*)d_in[16];
  const float* mb2 = (const float*)d_in[17];
  const float* uW1 = (const float*)d_in[18];
  const float* ub1 = (const float*)d_in[19];
  const float* uW2 = (const float*)d_in[20];
  const float* ub2 = (const float*)d_in[21];
  const float* u_g = (const float*)d_in[22];
  const float* u_b = (const float*)d_in[23];
  const float* gW1 = (const float*)d_in[24];
  const float* gb1 = (const float*)d_in[25];
  const float* gW2 = (const float*)d_in[26];
  const float* gb2 = (const float*)d_in[27];
  const float* g_g = (const float*)d_in[28];
  const float* g_b = (const float*)d_in[29];
  const int* src = (const int*)d_in[30];
  const int* tgt = (const int*)d_in[31];

  float* xs = (float*)d_out;                    // [NS,128]
  float* xt = xs + (size_t)NS_N * F;            // [NT,128]
  float* uo = xt + (size_t)NT_N * F;            // [1,128]

  const int NSP = ((NS_N + 63) / 64) * 64;      // 50048
  const int NTP = NT_N;                         // 256 (mult of 64)

  // workspace allocator (256B aligned)
  size_t off = 0;
  char* base = (char*)d_ws;
  auto alloc = [&](size_t bytes) -> char* {
    off = (off + 255) & ~(size_t)255;
    char* p = base + off;
    off += bytes;
    return p;
  };
  bf16_t* wt_eW1 = (bf16_t*)alloc((size_t)NLAYER * 512 * 512 * 2);
  bf16_t* wt_eW2 = (bf16_t*)alloc((size_t)NLAYER * 128 * 512 * 2);
  bf16_t* wt_mW1 = (bf16_t*)alloc((size_t)NLAYER * 256 * 256 * 2);
  bf16_t* wt_mW2 = (bf16_t*)alloc((size_t)NLAYER * 128 * 256 * 2);
  bf16_t* wt_uW1 = (bf16_t*)alloc((size_t)NLAYER * 128 * 256 * 2);
  bf16_t* wt_uW2 = (bf16_t*)alloc((size_t)NLAYER * 128 * 128 * 2);
  bf16_t* Hbuf   = (bf16_t*)alloc((size_t)E_N * 512 * 2);   // edge concat; reused as msg concat
  bf16_t* Hhid   = (bf16_t*)alloc((size_t)E_N * 512 * 2);   // edge hidden; reused as msg hidden
  float*  e_f32  = (float*) alloc((size_t)E_N * F * 4);     // edge out (pre+post BN)
  float*  msgf   = (float*) alloc((size_t)E_N * F * 4);     // message out
  float*  agg_s  = (float*) alloc((size_t)NSP * F * 4);
  float*  agg_t  = (float*) alloc((size_t)NTP * F * 4);
  bf16_t* Ucat   = (bf16_t*)alloc((size_t)NSP * 256 * 2);
  bf16_t* Uhid   = (bf16_t*)alloc((size_t)NSP * F * 2);
  float*  Uout   = (float*) alloc((size_t)NSP * F * 4);
  float*  cnt_s  = (float*) alloc((size_t)NSP * 4);
  float*  stats  = (float*) alloc(1024 * 4);  // [sum|sumsq|scale|shift|gsum_s|gsum_t]
  float* sum_c  = stats;        float* sumsq_c = stats + 128;
  float* scale_c = stats + 256; float* shift_c = stats + 384;
  float* gsum_s = stats + 512;  float* gsum_t  = stats + 640;

  const int TB = 256;
  bf16_t* Mcat = Hbuf;   // [E,256] aliases H (H dead after edge GEMM1)
  bf16_t* Mhid = Hhid;   // [E,256] aliases edge hidden (dead after edge GEMM2)

  // --- weight convert+transpose (all layers) ---
  for (int l = 0; l < NLAYER; ++l) {
    k_transpose_bf16<<<CEILDIV(512*512,TB),TB,0,stream>>>(eW1 + (size_t)l*512*512, wt_eW1 + (size_t)l*512*512, 512, 512);
    k_transpose_bf16<<<CEILDIV(512*128,TB),TB,0,stream>>>(eW2 + (size_t)l*512*128, wt_eW2 + (size_t)l*128*512, 512, 128);
    k_transpose_bf16<<<CEILDIV(256*256,TB),TB,0,stream>>>(mW1 + (size_t)l*256*256, wt_mW1 + (size_t)l*256*256, 256, 256);
    k_transpose_bf16<<<CEILDIV(256*128,TB),TB,0,stream>>>(mW2 + (size_t)l*256*128, wt_mW2 + (size_t)l*128*256, 256, 128);
    k_transpose_bf16<<<CEILDIV(256*128,TB),TB,0,stream>>>(uW1 + (size_t)l*256*128, wt_uW1 + (size_t)l*128*256, 256, 128);
    k_transpose_bf16<<<CEILDIV(128*128,TB),TB,0,stream>>>(uW2 + (size_t)l*128*128, wt_uW2 + (size_t)l*128*128, 128, 128);
  }

  // --- encoders + u init + edge counts (scatter-mean denominator) ---
  k_enc_s<<<CEILDIV(NS_N*F,TB),TB,0,stream>>>(x_s_raw, enc_s_W, enc_s_b, xs);
  k_enc_t<<<CEILDIV(NT_N*F,TB),TB,0,stream>>>(x_t_raw, enc_t_W, enc_t_b, xt);
  k_copy_u<<<1,128,0,stream>>>(x_u, uo);
  hipMemsetAsync(cnt_s, 0, (size_t)NSP * 4, stream);
  k_count<<<CEILDIV(E_N,TB),TB,0,stream>>>(src, cnt_s);

  for (int l = 0; l < NLAYER; ++l) {
    // ---------------- EdgeModel ----------------
    k_edge_concat<<<CEILDIV(E_N*512,TB),TB,0,stream>>>(xs, xt, edge_attr, uo, src, tgt, Hbuf);
    wmma_gemm_bf16<true,1><<<dim3(E_N/64,4),TB,0,stream>>>(Hbuf, wt_eW1 + (size_t)l*512*512, eb1 + l*512, Hhid, E_N, 512, 512);
    wmma_gemm_bf16<false,0><<<dim3(E_N/64,1),TB,0,stream>>>(Hhid, wt_eW2 + (size_t)l*128*512, eb2 + l*128, e_f32, E_N, 128, 512);
    hipMemsetAsync(sum_c, 0, 256 * 4, stream);
    k_colstats<<<512,128,0,stream>>>(e_f32, E_N, sum_c, sumsq_c);
    k_bn_final<<<1,128,0,stream>>>(sum_c, sumsq_c, e_g + l*F, e_b + l*F, E_N, scale_c, shift_c);
    k_bn_apply<<<CEILDIV(E_N*F,TB),TB,0,stream>>>(e_f32, scale_c, shift_c, E_N, e_f32);

    // ---------------- messages onto sources (mean over src) ----------------
    k_msg_concat<<<CEILDIV(E_N*256,TB),TB,0,stream>>>(xt, tgt, e_f32, Mcat);
    wmma_gemm_bf16<true,1><<<dim3(E_N/64,2),TB,0,stream>>>(Mcat, wt_mW1 + (size_t)l*256*256, mb1 + l*256, Mhid, E_N, 256, 256);
    wmma_gemm_bf16<false,0><<<dim3(E_N/64,1),TB,0,stream>>>(Mhid, wt_mW2 + (size_t)l*128*256, mb2 + l*128, msgf, E_N, 128, 256);
    hipMemsetAsync(agg_s, 0, (size_t)NSP * F * 4, stream);
    k_scatter_add<<<CEILDIV(E_N*F,TB),TB,0,stream>>>(msgf, src, agg_s);

    // ---------------- messages onto targets (sum over tgt; uses OLD x_s) ---
    k_msg_concat<<<CEILDIV(E_N*256,TB),TB,0,stream>>>(xs, src, e_f32, Mcat);
    wmma_gemm_bf16<true,1><<<dim3(E_N/64,2),TB,0,stream>>>(Mcat, wt_mW1 + (size_t)l*256*256, mb1 + l*256, Mhid, E_N, 256, 256);
    wmma_gemm_bf16<false,0><<<dim3(E_N/64,1),TB,0,stream>>>(Mhid, wt_mW2 + (size_t)l*128*256, mb2 + l*128, msgf, E_N, 128, 256);
    hipMemsetAsync(agg_t, 0, (size_t)NTP * F * 4, stream);
    k_scatter_add<<<CEILDIV(E_N*F,TB),TB,0,stream>>>(msgf, tgt, agg_t);

    // ---------------- node update: sources -------------------------------
    k_upd_concat<<<CEILDIV(NS_N*256,TB),TB,0,stream>>>(xs, agg_s, cnt_s, NS_N, Ucat);
    wmma_gemm_bf16<true,1><<<dim3(CEILDIV(NS_N,64),1),TB,0,stream>>>(Ucat, wt_uW1 + (size_t)l*128*256, ub1 + l*F, Uhid, NS_N, 128, 256);
    wmma_gemm_bf16<false,0><<<dim3(CEILDIV(NS_N,64),1),TB,0,stream>>>(Uhid, wt_uW2 + (size_t)l*128*128, ub2 + l*F, Uout, NS_N, 128, 128);
    hipMemsetAsync(sum_c, 0, 256 * 4, stream);
    k_colstats<<<512,128,0,stream>>>(Uout, NS_N, sum_c, sumsq_c);
    k_bn_final<<<1,128,0,stream>>>(sum_c, sumsq_c, u_g + l*F, u_b + l*F, NS_N, scale_c, shift_c);
    k_bn_apply<<<CEILDIV(NS_N*F,TB),TB,0,stream>>>(Uout, scale_c, shift_c, NS_N, xs);

    // ---------------- node update: targets -------------------------------
    k_upd_concat<<<CEILDIV(NT_N*256,TB),TB,0,stream>>>(xt, agg_t, nullptr, NT_N, Ucat);
    wmma_gemm_bf16<true,1><<<dim3(CEILDIV(NT_N,64),1),TB,0,stream>>>(Ucat, wt_uW1 + (size_t)l*128*256, ub1 + l*F, Uhid, NT_N, 128, 256);
    wmma_gemm_bf16<false,0><<<dim3(CEILDIV(NT_N,64),1),TB,0,stream>>>(Uhid, wt_uW2 + (size_t)l*128*128, ub2 + l*F, Uout, NT_N, 128, 128);
    hipMemsetAsync(sum_c, 0, 256 * 4, stream);
    k_colstats<<<256,128,0,stream>>>(Uout, NT_N, sum_c, sumsq_c);
    k_bn_final<<<1,128,0,stream>>>(sum_c, sumsq_c, u_g + l*F, u_b + l*F, NT_N, scale_c, shift_c);
    k_bn_apply<<<CEILDIV(NT_N*F,TB),TB,0,stream>>>(Uout, scale_c, shift_c, NT_N, xt);

    // ---------------- GlobalModel -----------------------------------------
    hipMemsetAsync(gsum_s, 0, 256 * 4, stream);
    k_colstats<<<512,128,0,stream>>>(xs, NS_N, gsum_s, nullptr);
    k_colstats<<<256,128,0,stream>>>(xt, NT_N, gsum_t, nullptr);
    k_global<<<1,384,0,stream>>>(uo, gsum_s, gsum_t,
                                 gW1 + (size_t)l*384*384, gb1 + l*384,
                                 gW2 + (size_t)l*384*128, gb2 + l*128,
                                 g_g + l*F, g_b + l*F, uo);
  }
}
